// RebuiltMetabolismGenerator_34900904247640
// MI455X (gfx1250) — matrix operand order repaired and checked by reading, not compiled, not run on hardware
//
#include <hip/hip_runtime.h>
#include <hip/hip_bf16.h>

// ---------------------------------------------------------------------------
// Problem constants (from reference)
// ---------------------------------------------------------------------------
#define HH        4
#define CC        128
#define MAX_NODES 50
#define NF        64
#define N_NODES   10000
#define N_EDGES   160000
#define N_GRAPHS  64
#define LAT       64
#define HID       128
#define NPAIR     1225            // 50*49/2
#define E_TOT     (N_EDGES + N_NODES)

typedef __attribute__((ext_vector_type(16))) _Float16 v16h;
typedef __attribute__((ext_vector_type(8)))  _Float16 v8h;
typedef __attribute__((ext_vector_type(4)))  _Float16 v4h;
typedef __attribute__((ext_vector_type(8)))  float    v8f;
typedef __attribute__((ext_vector_type(4)))  float    v4f;

// ---------------------------------------------------------------------------
// Generic WMMA GEMM:  C[M,N] = act(A[M,K](lda) @ B[K,N](ldb) + bias)
// fp32 in/out, f16 multiply + f32 accumulate via v_wmma_f32_16x16x32_f16.
// Block tile 128(M) x 64(N), 8 wave32 waves in a 4x2 grid; each wave owns a
// 32x32 strip -> 2 A-frags x 2 B-frags = 4 WMMAs per K-step (high
// WMMA:DS-op ratio; WMMA co-executes with the LDS fragment gathers).
// Full interior tiles take an unconditional float4 load path; edge tiles use
// branch-free clamped loads (select, not exec-divergence).
// act: 0 = none, 1 = relu, 2 = sigmoid
// ---------------------------------------------------------------------------
__global__ __launch_bounds__(256) void gemm_wmma_kernel(
    const float* __restrict__ A, int lda,
    const float* __restrict__ B, int ldb,
    const float* __restrict__ bias,
    float* __restrict__ C, int ldc,
    int M, int N, int K, int act)
{
    __shared__ _Float16 As[128][40];  // 128 x 32 tile (+8 pad, rows 16B aligned)
    __shared__ _Float16 Bs[32][72];   //  32 x 64 tile (+8 pad, rows 16B aligned)

    const int tid  = threadIdx.x;
    const int wave = tid >> 5;
    const int lane = tid & 31;
    const int wr   = wave >> 1;       // 0..3  (wave row: 32 output rows each)
    const int wc   = wave & 1;        // 0..1  (wave col: 32 output cols each)
    const int row0 = blockIdx.y * 128;
    const int col0 = blockIdx.x * 64;

    v8f acc[2][2];
    #pragma unroll
    for (int i = 0; i < 2; ++i)
        #pragma unroll
        for (int j = 0; j < 2; ++j)
            acc[i][j] = (v8f){0.f, 0.f, 0.f, 0.f, 0.f, 0.f, 0.f, 0.f};

    const bool fullM = (row0 + 128 <= M);
    const bool fullN = (col0 + 64 <= N);

    for (int k0 = 0; k0 < K; k0 += 32) {
        const bool fullK = (k0 + 32 <= K);

        // ---- stage A tile: 128 rows x 32 k (4096 floats, 1024 float4) ----
        if (fullM && fullK) {
            #pragma unroll
            for (int t = 0; t < 4; ++t) {
                int idx = tid + t * 256;           // 0..1023
                int r  = idx >> 3;                 // 8 float4 per row
                int kq = (idx & 7) * 4;
                v4f f = *(const v4f*)(A + (size_t)(row0 + r) * lda + k0 + kq);
                v4h hq = {(_Float16)f[0], (_Float16)f[1],
                          (_Float16)f[2], (_Float16)f[3]};
                *(v4h*)&As[r][kq] = hq;
            }
        } else {
            for (int i = tid; i < 128 * 32; i += 256) {
                int r  = i >> 5, kk = i & 31;
                int gr = row0 + r, gk = k0 + kk;
                int grc = gr < M ? gr : M - 1;
                int gkc = gk < K ? gk : K - 1;
                float v = A[(size_t)grc * lda + gkc];       // always in-range
                As[r][kk] = (_Float16)((gr < M && gk < K) ? v : 0.f);
            }
        }

        // ---- stage B tile: 32 k x 64 n (2048 floats, 512 float4) ----
        if (fullN && fullK) {
            #pragma unroll
            for (int t = 0; t < 2; ++t) {
                int idx = tid + t * 256;
                int kr = idx >> 4;                 // 16 float4 per row
                int nq = (idx & 15) * 4;
                v4f f = *(const v4f*)(B + (size_t)(k0 + kr) * ldb + col0 + nq);
                v4h hq = {(_Float16)f[0], (_Float16)f[1],
                          (_Float16)f[2], (_Float16)f[3]};
                *(v4h*)&Bs[kr][nq] = hq;
            }
        } else {
            for (int i = tid; i < 32 * 64; i += 256) {
                int kk = i >> 6, c = i & 63;
                int gk = k0 + kk, gc = col0 + c;
                int gkc = gk < K ? gk : K - 1;
                int gcc = gc < N ? gc : N - 1;
                float v = B[(size_t)gkc * ldb + gcc];
                Bs[kk][c] = (_Float16)((gk < K && gc < N) ? v : 0.f);
            }
        }
        __syncthreads();

        // ---- A fragments (ISA 7.12.2, 16-bit A 16x32, wave32) ----
        // lanes 0-15: row=lane,    h[0..7]=K0..7,  h[8..15]=K16..23
        // lanes16-31: row=lane-16, h[0..7]=K8..15, h[8..15]=K24..31
        const int kb8 = (lane >> 4) * 8;
        v16h afrag[2];
        #pragma unroll
        for (int i2 = 0; i2 < 2; ++i2) {
            const int arow = wr * 32 + i2 * 16 + (lane & 15);
            v8h a0 = *(const v8h*)&As[arow][kb8];        // 16B ds_load
            v8h a1 = *(const v8h*)&As[arow][kb8 + 16];   // 16B ds_load
            #pragma unroll
            for (int i = 0; i < 8; ++i) { afrag[i2][i] = a0[i]; afrag[i2][i + 8] = a1[i]; }
        }

        // ---- B fragments (16-bit B 32x16): same K order as A (lanes 0-15 K0..7,16..23; 16-31 K8..15,24..31)
        const int kb16 = (lane >> 4) * 8;
        v16h bfrag[2];
        #pragma unroll
        for (int j2 = 0; j2 < 2; ++j2) {
            const int bcol = wc * 32 + j2 * 16 + (lane & 15);
            #pragma unroll
            for (int i = 0; i < 16; ++i) bfrag[j2][i] = Bs[kb16 + (i & 7) + ((i >> 3) << 4)][bcol];
        }

        #pragma unroll
        for (int i2 = 0; i2 < 2; ++i2)
            #pragma unroll
            for (int j2 = 0; j2 < 2; ++j2)
                acc[i2][j2] = __builtin_amdgcn_wmma_f32_16x16x32_f16(
                    false, afrag[i2], false, bfrag[j2],
                    (short)0, acc[i2][j2], false, false);

        __syncthreads();
    }

    // ---- epilogue (C/D layout: lanes 0-15 N=lane,M=r; 16-31 N=lane-16,M=8+r)
    #pragma unroll
    for (int i2 = 0; i2 < 2; ++i2) {
        const int m0 = row0 + wr * 32 + i2 * 16 + (lane >> 4) * 8;
        #pragma unroll
        for (int j2 = 0; j2 < 2; ++j2) {
            const int n = col0 + wc * 32 + j2 * 16 + (lane & 15);
            float bv = (bias != nullptr && n < N) ? bias[n] : 0.f;
            #pragma unroll
            for (int r = 0; r < 8; ++r) {
                int m = m0 + r;
                if (m < M && n < N) {
                    float v = acc[i2][j2][r] + bv;
                    if (act == 1)      v = v > 0.f ? v : 0.f;
                    else if (act == 2) v = 1.f / (1.f + __expf(-v));
                    C[(size_t)m * ldc + n] = v;
                }
            }
        }
    }
}

// ---------------------------------------------------------------------------
// Small elementwise / scatter kernels
// ---------------------------------------------------------------------------
__device__ inline void atomicMaxF(float* addr, float val) {
    unsigned int* ai = (unsigned int*)addr;
    unsigned int old = *ai;
    while (__uint_as_float(old) < val) {
        unsigned int assumed = old;
        old = atomicCAS(ai, assumed, __float_as_uint(val));
        if (old == assumed) break;
    }
}

__global__ void k_fill(float* p, float v, int n) {
    int t = blockIdx.x * blockDim.x + threadIdx.x;
    if (t < n) p[t] = v;
}

// es/ed attention scores: es[n,h] = sum_c xh[n,h,c]*a_src[h,c]
__global__ void k_scores(const float* __restrict__ xh,
                         const float* __restrict__ a_s,
                         const float* __restrict__ a_d,
                         float* __restrict__ es, float* __restrict__ ed) {
    int t = blockIdx.x * blockDim.x + threadIdx.x;   // n*HH + h
    if (t >= N_NODES * HH) return;
    int hh = t & 3, n = t >> 2;
    const float* row = xh + (size_t)n * (HH * CC) + hh * CC;
    const float* w1  = a_s + hh * CC;
    const float* w2  = a_d + hh * CC;
    float s1 = 0.f, s2 = 0.f;
    for (int c = 0; c < CC; ++c) { float x = row[c]; s1 += x * w1[c]; s2 += x * w2[c]; }
    es[t] = s1; ed[t] = s2;
}

// pass 1: e = leaky_relu(es[src]+ed[dst]); segment_max into m (CAS-max)
__global__ void k_edge1(const int* __restrict__ ei,
                        const float* __restrict__ es, const float* __restrict__ ed,
                        float* __restrict__ val, float* __restrict__ m) {
    int t = blockIdx.x * blockDim.x + threadIdx.x;   // e*HH + h
    if (t >= E_TOT * HH) return;
    int hh = t & 3, e = t >> 2;
    int s, d;
    if (e < N_EDGES) { s = ei[e]; d = ei[N_EDGES + e]; }
    else             { s = d = e - N_EDGES; }
    float v = es[s * HH + hh] + ed[d * HH + hh];
    if (v < 0.f) v *= 0.2f;
    val[t] = v;
    atomicMaxF(&m[d * HH + hh], v);
}

// pass 2: ex = exp(e - m[dst]); den[dst] += ex ; store ex in place of val
__global__ void k_edge2(const int* __restrict__ ei,
                        float* __restrict__ val, const float* __restrict__ m,
                        float* __restrict__ den) {
    int t = blockIdx.x * blockDim.x + threadIdx.x;
    if (t >= E_TOT * HH) return;
    int hh = t & 3, e = t >> 2;
    int d = (e < N_EDGES) ? ei[N_EDGES + e] : (e - N_EDGES);
    float ex = __expf(val[t] - m[d * HH + hh]);
    val[t] = ex;
    atomicAdd(&den[d * HH + hh], ex);
}

// pass 3: out[dst,h,c] += xh[src,h,c] * (ex/den[dst,h])
__global__ void k_edge3(const int* __restrict__ ei,
                        const float* __restrict__ xh,
                        const float* __restrict__ val, const float* __restrict__ den,
                        float* __restrict__ gout) {
    int t = blockIdx.x * blockDim.x + threadIdx.x;   // e*CC + c
    if (t >= E_TOT * CC) return;
    int c = t & (CC - 1), e = t >> 7;
    int s, d;
    if (e < N_EDGES) { s = ei[e]; d = ei[N_EDGES + e]; }
    else             { s = d = e - N_EDGES; }
    #pragma unroll
    for (int hh = 0; hh < HH; ++hh) {
        float alpha = val[e * HH + hh] / den[d * HH + hh];
        atomicAdd(&gout[(size_t)d * (HH * CC) + hh * CC + c],
                  xh[(size_t)s * (HH * CC) + hh * CC + c] * alpha);
    }
}

// h += relu(mean_heads(gout) + bg)
__global__ void k_combine(const float* __restrict__ gout,
                          const float* __restrict__ bg, float* __restrict__ h) {
    int t = blockIdx.x * blockDim.x + threadIdx.x;   // n*HID + c
    if (t >= N_NODES * HID) return;
    int c = t & (HID - 1), n = t >> 7;
    const float* g = gout + (size_t)n * (HH * CC);
    float mv = 0.25f * (g[c] + g[CC + c] + g[2 * CC + c] + g[3 * CC + c]);
    float v = mv + bg[c];
    if (v < 0.f) v = 0.f;
    h[t] += v;
}

__global__ void k_pool_sum(const float* __restrict__ h, const int* __restrict__ batch,
                           float* __restrict__ sums) {
    int t = blockIdx.x * blockDim.x + threadIdx.x;
    if (t >= N_NODES * HID) return;
    int c = t & (HID - 1), n = t >> 7;
    atomicAdd(&sums[batch[n] * HID + c], h[t]);
}

__global__ void k_pool_cnt(const int* __restrict__ batch, float* __restrict__ counts) {
    int t = blockIdx.x * blockDim.x + threadIdx.x;
    if (t >= N_NODES) return;
    atomicAdd(&counts[batch[t]], 1.f);
}

__global__ void k_pool_div(const float* __restrict__ sums, const float* __restrict__ counts,
                           float* __restrict__ pr) {
    int t = blockIdx.x * blockDim.x + threadIdx.x;
    if (t >= N_GRAPHS * HID) return;
    int b = t >> 7;
    pr[t] = sums[t] / fmaxf(counts[b], 1.f);
}

__global__ void k_reparam(const float* __restrict__ mu, const float* __restrict__ lv,
                          const float* __restrict__ eps, float* __restrict__ z) {
    int t = blockIdx.x * blockDim.x + threadIdx.x;
    if (t >= N_GRAPHS * LAT) return;
    z[t] = mu[t] + eps[t] * __expf(0.5f * lv[t]);
}

__global__ void k_sigmoid(const float* __restrict__ in, float* __restrict__ out, int n) {
    int t = blockIdx.x * blockDim.x + threadIdx.x;
    if (t < n) out[t] = 1.f / (1.f + __expf(-in[t]));
}

__global__ void k_pairs(int* __restrict__ iu, int* __restrict__ ju) {
    if (blockIdx.x == 0 && threadIdx.x == 0) {
        int p = 0;
        for (int i = 0; i < MAX_NODES; ++i)
            for (int j = i + 1; j < MAX_NODES; ++j) { iu[p] = i; ju[p] = j; ++p; }
    }
}

// Z1[b*P+p, :] = [ z[b,0:64] | nodes[b, iu[p], :] | nodes[b, ju[p], :] ]
__global__ void k_edgefeat(const float* __restrict__ z, const float* __restrict__ nodes,
                           const int* __restrict__ iu, const int* __restrict__ ju,
                           float* __restrict__ Z1) {
    int t = blockIdx.x * blockDim.x + threadIdx.x;   // (b*P+p)*192 + c
    if (t >= N_GRAPHS * NPAIR * 192) return;
    int c  = t % 192;
    int bp = t / 192;
    int p  = bp % NPAIR;
    int b  = bp / NPAIR;
    float v;
    if (c < 64)       v = z[b * LAT + c];
    else if (c < 128) v = nodes[(size_t)b * (MAX_NODES * NF) + iu[p] * NF + (c - 64)];
    else              v = nodes[(size_t)b * (MAX_NODES * NF) + ju[p] * NF + (c - 128)];
    Z1[t] = v;
}

__global__ void k_softmax6(const float* __restrict__ logits, float* __restrict__ out) {
    int t = blockIdx.x * blockDim.x + threadIdx.x;   // row over B*P
    if (t >= N_GRAPHS * NPAIR) return;
    const float* r = logits + (size_t)t * 6;
    float mx = r[0];
    #pragma unroll
    for (int k = 1; k < 6; ++k) mx = fmaxf(mx, r[k]);
    float s = 0.f, e[6];
    #pragma unroll
    for (int k = 0; k < 6; ++k) { e[k] = __expf(r[k] - mx); s += e[k]; }
    float inv = 1.f / s;
    #pragma unroll
    for (int k = 0; k < 6; ++k) out[(size_t)t * 6 + k] = e[k] * inv;
}

// ---------------------------------------------------------------------------
// Host side
// ---------------------------------------------------------------------------
static inline int cdiv(int a, int b) { return (a + b - 1) / b; }

static void gemm(hipStream_t s, const float* A, int lda, const float* B, int ldb,
                 const float* bias, float* C, int ldc, int M, int N, int K, int act) {
    dim3 grid(cdiv(N, 64), cdiv(M, 128));
    gemm_wmma_kernel<<<grid, 256, 0, s>>>(A, lda, B, ldb, bias, C, ldc, M, N, K, act);
}

static void fill(hipStream_t s, float* p, float v, int n) {
    k_fill<<<cdiv(n, 256), 256, 0, s>>>(p, v, n);
}

extern "C" void kernel_launch(void* const* d_in, const int* in_sizes, int n_in,
                              void* d_out, int out_size, void* d_ws, size_t ws_size,
                              hipStream_t stream) {
    (void)in_sizes; (void)n_in; (void)out_size; (void)ws_size;

    // ---- inputs (setup_inputs dict order) ----
    const float* x     = (const float*)d_in[0];
    const int*   ei    = (const int*)  d_in[1];
    const int*   batch = (const int*)  d_in[2];
    const float* eps   = (const float*)d_in[3];
    const float* We    = (const float*)d_in[4];
    const float* be    = (const float*)d_in[5];
    const float* Wg    = (const float*)d_in[6];   // (3,128,512)
    const float* ag_s  = (const float*)d_in[7];   // (3,4,128)
    const float* ag_d  = (const float*)d_in[8];
    const float* bg    = (const float*)d_in[9];   // (3,128)
    const float* Wa1   = (const float*)d_in[10];
    const float* ba1   = (const float*)d_in[11];
    const float* Wa2   = (const float*)d_in[12];
    const float* ba2   = (const float*)d_in[13];
    const float* Wmu   = (const float*)d_in[14];
    const float* bmu   = (const float*)d_in[15];
    const float* Wlv   = (const float*)d_in[16];
    const float* blv   = (const float*)d_in[17];
    const float* Wn1   = (const float*)d_in[18];
    const float* bn1   = (const float*)d_in[19];
    const float* Wn2   = (const float*)d_in[20];
    const float* bn2   = (const float*)d_in[21];
    const float* Wn3   = (const float*)d_in[22];
    const float* bn3   = (const float*)d_in[23];
    const float* We1   = (const float*)d_in[24];
    const float* be1   = (const float*)d_in[25];
    const float* We2   = (const float*)d_in[26];
    const float* be2   = (const float*)d_in[27];
    const float* We3   = (const float*)d_in[28];
    const float* be3   = (const float*)d_in[29];
    const float* Wr1   = (const float*)d_in[30];
    const float* br1   = (const float*)d_in[31];
    const float* Wr2   = (const float*)d_in[32];
    const float* br2   = (const float*)d_in[33];

    // ---- output slices (concatenated, fp32) ----
    float* out   = (float*)d_out;
    float* o_mu  = out;                                   // 64*64
    float* o_lv  = o_mu + N_GRAPHS * LAT;                 // 64*64
    float* o_z   = o_lv + N_GRAPHS * LAT;                 // 64*64
    float* o_pr  = o_z  + N_GRAPHS * LAT;                 // 64*128
    float* o_np  = o_pr + N_GRAPHS * HID;                 // 64*50*64
    float* o_ep  = o_np + N_GRAPHS * MAX_NODES * NF;      // 64*1225
    float* o_rt  = o_ep + N_GRAPHS * NPAIR;               // 64*1225*6

    // ---- workspace layout ----
    float* W = (float*)d_ws;
    size_t o = 0;
    auto alloc = [&](size_t n) { float* p = W + o; o += n; return p; };
    float* h      = alloc((size_t)N_NODES * HID);
    float* xh     = alloc((size_t)N_NODES * HH * CC);
    float* gout   = alloc((size_t)N_NODES * HH * CC);
    float* es     = alloc((size_t)N_NODES * HH);
    float* ed     = alloc((size_t)N_NODES * HH);
    float* mseg   = alloc((size_t)N_NODES * HH);
    float* den    = alloc((size_t)N_NODES * HH);
    float* val    = alloc((size_t)E_TOT * HH);
    float* sums   = alloc((size_t)N_GRAPHS * HID);
    float* counts = alloc((size_t)N_GRAPHS);
    float* prpool = alloc((size_t)N_GRAPHS * HID);
    float* a1     = alloc((size_t)N_GRAPHS * 2 * HID);
    float* g1     = alloc((size_t)N_GRAPHS * 256);
    float* g2     = alloc((size_t)N_GRAPHS * 512);
    float* nodes  = alloc((size_t)N_GRAPHS * MAX_NODES * NF);
    int*   iu     = (int*)alloc(NPAIR + 7);
    int*   ju     = (int*)alloc(NPAIR + 7);
    float* Z1     = alloc((size_t)N_GRAPHS * NPAIR * 192);
    float* eh1    = alloc((size_t)N_GRAPHS * NPAIR * 128);
    float* eh2    = alloc((size_t)N_GRAPHS * NPAIR * 64);
    float* rh     = alloc((size_t)N_GRAPHS * NPAIR * 64);
    float* rlog   = alloc((size_t)N_GRAPHS * NPAIR * 6);

    // ---- encoder: h = relu(x @ We + be) ----
    gemm(stream, x, NF, We, HID, be, h, HID, N_NODES, HID, NF, /*relu*/1);

    // ---- 3 GAT layers ----
    for (int l = 0; l < 3; ++l) {
        gemm(stream, h, HID, Wg + (size_t)l * HID * (HH * CC), HH * CC,
             nullptr, xh, HH * CC, N_NODES, HH * CC, HID, 0);

        k_scores<<<cdiv(N_NODES * HH, 256), 256, 0, stream>>>(
            xh, ag_s + l * HH * CC, ag_d + l * HH * CC, es, ed);

        fill(stream, mseg, -1e30f, N_NODES * HH);
        fill(stream, den, 0.f, N_NODES * HH);
        fill(stream, gout, 0.f, N_NODES * HH * CC);

        k_edge1<<<cdiv(E_TOT * HH, 256), 256, 0, stream>>>(ei, es, ed, val, mseg);
        k_edge2<<<cdiv(E_TOT * HH, 256), 256, 0, stream>>>(ei, val, mseg, den);
        k_edge3<<<cdiv(E_TOT * CC, 256), 256, 0, stream>>>(ei, xh, val, den, gout);
        k_combine<<<cdiv(N_NODES * HID, 256), 256, 0, stream>>>(gout, bg + l * CC, h);
    }

    // ---- mean pooling ----
    fill(stream, sums, 0.f, N_GRAPHS * HID);
    fill(stream, counts, 0.f, N_GRAPHS);
    k_pool_sum<<<cdiv(N_NODES * HID, 256), 256, 0, stream>>>(h, batch, sums);
    k_pool_cnt<<<cdiv(N_NODES, 256), 256, 0, stream>>>(batch, counts);
    k_pool_div<<<cdiv(N_GRAPHS * HID, 256), 256, 0, stream>>>(sums, counts, prpool);

    // ---- graph head: pr, mu, logvar, z ----
    gemm(stream, prpool, HID, Wa1, 2 * HID, ba1, a1, 2 * HID, N_GRAPHS, 2 * HID, HID, 1);
    gemm(stream, a1, 2 * HID, Wa2, HID, ba2, o_pr, HID, N_GRAPHS, HID, 2 * HID, 0);
    gemm(stream, o_pr, HID, Wmu, LAT, bmu, o_mu, LAT, N_GRAPHS, LAT, HID, 0);
    gemm(stream, o_pr, HID, Wlv, LAT, blv, o_lv, LAT, N_GRAPHS, LAT, HID, 0);
    k_reparam<<<cdiv(N_GRAPHS * LAT, 256), 256, 0, stream>>>(o_mu, o_lv, eps, o_z);

    // ---- node decoder ----
    gemm(stream, o_z, LAT, Wn1, 256, bn1, g1, 256, N_GRAPHS, 256, LAT, 1);
    gemm(stream, g1, 256, Wn2, 512, bn2, g2, 512, N_GRAPHS, 512, 256, 1);
    gemm(stream, g2, 512, Wn3, MAX_NODES * NF, bn3, nodes, MAX_NODES * NF,
         N_GRAPHS, MAX_NODES * NF, 512, 0);
    k_sigmoid<<<cdiv(N_GRAPHS * MAX_NODES * NF, 256), 256, 0, stream>>>(
        nodes, o_np, N_GRAPHS * MAX_NODES * NF);

    // ---- pair features: [z | n_i | n_j] ----
    k_pairs<<<1, 32, 0, stream>>>(iu, ju);
    k_edgefeat<<<cdiv(N_GRAPHS * NPAIR * 192, 256), 256, 0, stream>>>(
        o_z, nodes, iu, ju, Z1);

    // ---- edge MLP -> edge_probs ----
    const int BP = N_GRAPHS * NPAIR;
    gemm(stream, Z1, 192, We1, 128, be1, eh1, 128, BP, 128, 192, 1);
    gemm(stream, eh1, 128, We2, 64, be2, eh2, 64, BP, 64, 128, 1);
    gemm(stream, eh2, 64, We3, 1, be3, o_ep, 1, BP, 1, 64, /*sigmoid*/2);

    // ---- reaction MLP -> softmax(6); A = columns 64..191 of Z1 (lda=192) ----
    gemm(stream, Z1 + 64, 192, Wr1, 64, br1, rh, 64, BP, 64, 128, 1);
    gemm(stream, rh, 64, Wr2, 6, br2, rlog, 6, BP, 6, 64, 0);
    k_softmax6<<<cdiv(BP, 256), 256, 0, stream>>>(rlog, o_rt);
}
